// TransformerBlock_85117661872174
// MI455X (gfx1250) — matrix operand order repaired and checked by reading, not compiled, not run on hardware
//
#include <hip/hip_runtime.h>
#include <hip/hip_bf16.h>
#include <stdint.h>

#define DEVI __device__ __forceinline__

typedef __attribute__((ext_vector_type(16))) __bf16        bf16x16;
typedef __attribute__((ext_vector_type(8)))  float         f32x8;
typedef __attribute__((ext_vector_type(4)))  unsigned int  u32x4;

// exact pointee type of the async-LDS builtins: gcc-vector of 4 ints
typedef int i32x4v __attribute__((vector_size(16)));
typedef __attribute__((address_space(1))) i32x4v gvec128;   // global
typedef __attribute__((address_space(3))) i32x4v lvec128;   // LDS

namespace cfg {
constexpr int B   = 8;
constexpr int T   = 1024;
constexpr int C   = 768;
constexpr int H   = 12;
constexpr int D   = 64;     // head dim
constexpr int C3  = 3 * C;  // 2304
constexpr int C4  = 4 * C;  // 3072
constexpr int ROWS = B * T; // 8192
}

union FragU { bf16x16 v; u32x4 q[2]; };

// ---------------------------------------------------------------------------
// CDNA5 async global->LDS copy (16B per lane, ASYNCcnt-tracked) + wait
// ---------------------------------------------------------------------------
DEVI void async_cp16(const __bf16* gsrc, __bf16* ldst) {
#if defined(__has_builtin) && __has_builtin(__builtin_amdgcn_global_load_async_to_lds_b128)
  __builtin_amdgcn_global_load_async_to_lds_b128(
      (gvec128*)(uintptr_t)gsrc,
      (lvec128*)(unsigned)(uintptr_t)ldst, 0, 0);
#else
  asm volatile("global_load_async_to_lds_b128 %0, %1, off"
               :: "v"((unsigned)(uintptr_t)ldst),
                  "v"((unsigned long long)(uintptr_t)gsrc)
               : "memory");
#endif
}

template <int N> DEVI void wait_async() {
#if defined(__has_builtin) && __has_builtin(__builtin_amdgcn_s_wait_asynccnt)
  __builtin_amdgcn_s_wait_asynccnt(N);
#else
  asm volatile("s_wait_asynccnt %0" :: "i"(N) : "memory");
#endif
}

// ---------------------------------------------------------------------------
// WMMA wrapper: D = A(16x32 bf16) * B(32x16 bf16) + C(16x16 f32)
// ---------------------------------------------------------------------------
DEVI f32x8 wmma_bf16(bf16x16 a, bf16x16 b, f32x8 c) {
  return __builtin_amdgcn_wmma_f32_16x16x32_bf16(
      /*neg_a=*/false, a, /*neg_b=*/false, b,
      /*c_mod=*/(short)0, c, /*reuse_a=*/false, /*reuse_b=*/false);
}

// ---------------------------------------------------------------------------
// Fragment loaders (wave32; contiguous k order used by this code, same for A and B)
// A 16x32 bf16: lane m = lane&15, half = lane>>4.
//   lanes 0-15 hold k=0..15, lanes 16-31 hold k=16..31 (contiguous)
// B 32x16 bf16: lane n = lane&15; lanes 0-15 hold k=0..15, lanes 16-31 k=16..31
// ---------------------------------------------------------------------------
DEVI bf16x16 load_frag_a_lds(const __bf16* lds, int row0, int ld, int k0, int lane) {
  int m = lane & 15, half = lane >> 4;
  const __bf16* p = lds + (row0 + m) * ld + k0 + half * 16;
  FragU f;
  f.q[0] = *(const u32x4*)(p);
  f.q[1] = *(const u32x4*)(p + 8);
  return f.v;
}

DEVI bf16x16 load_frag_a_rows(const __bf16* base, long ldRow, int k0, int lane) {
  int m = lane & 15, half = lane >> 4;
  const __bf16* p = base + (long)m * ldRow + k0 + half * 16;
  FragU f;
  f.q[0] = *(const u32x4*)(p);
  f.q[1] = *(const u32x4*)(p + 8);
  return f.v;
}

// B fragment from LDS holding W^T tile: lds[n][k]
DEVI bf16x16 load_frag_b_ldsT(const __bf16* lds, int col0, int ld, int k0, int lane) {
  int n = lane & 15, kb = (lane >> 4) * 16;
  const __bf16* p = lds + (col0 + n) * ld + k0 + kb;
  FragU f;
  f.q[0] = *(const u32x4*)(p);
  f.q[1] = *(const u32x4*)(p + 8);
  return f.v;
}

// B fragment where n indexes memory rows (K^T for attention scores)
DEVI bf16x16 load_frag_b_rows(const __bf16* base, long ldRow, int d0, int lane) {
  const __bf16* p = base + (long)(lane & 15) * ldRow + d0 + ((lane >> 4) * 16);
  FragU f;
  f.q[0] = *(const u32x4*)(p);
  f.q[1] = *(const u32x4*)(p + 8);
  return f.v;
}

// B fragment from pre-transposed V (vT[d][t]): lane n -> d, kdim -> t.
DEVI bf16x16 load_frag_b_cols(const __bf16* base, long ldRow, int d0, int t0, int lane) {
  const __bf16* p = base + (long)(d0 + (lane & 15)) * ldRow + t0 + ((lane >> 4) * 16);
  FragU f;
  f.q[0] = *(const u32x4*)(p);
  f.q[1] = *(const u32x4*)(p + 8);
  return f.v;
}

DEVI float gelu_tanh(float x) {
  float x3 = x * x * x;
  return 0.5f * x * (1.0f + tanhf(0.79788456080286535588f * (x + 0.044715f * x3)));
}

// ---------------------------------------------------------------------------
// Weight prepass: wT[n][k] = (bf16) w[k][n]   (w is [K][N] row-major)
// ---------------------------------------------------------------------------
__global__ void transpose_w_bf16(const float* __restrict__ w, __bf16* __restrict__ wT,
                                 int K, int N) {
  long total = (long)K * N;
  long idx = (long)blockIdx.x * blockDim.x + threadIdx.x;
  if (idx >= total) return;
  int k = (int)(idx % K);
  int n = (int)(idx / K);
  wT[idx] = (__bf16)w[(long)k * N + n];
}

// ---------------------------------------------------------------------------
// LayerNorm (torch unbiased var, ddof=1), f32 in -> bf16 out
// ---------------------------------------------------------------------------
__global__ __launch_bounds__(256) void layernorm_bf16(
    const float* __restrict__ x, const float* __restrict__ w,
    const float* __restrict__ b, __bf16* __restrict__ out) {
  using namespace cfg;
  int row = blockIdx.x;
  int tid = threadIdx.x, lane = tid & 31, wid = tid >> 5;
  const float* xr = x + (long)row * C;

  float s = 0.f, s2 = 0.f;
  for (int i = tid; i < C; i += 256) {
    float v = xr[i];
    s += v; s2 += v * v;
  }
  #pragma unroll
  for (int i = 1; i < 32; i <<= 1) {
    s  += __shfl_xor(s,  i, 32);
    s2 += __shfl_xor(s2, i, 32);
  }
  __shared__ float rs[8], rs2[8];
  if (lane == 0) { rs[wid] = s; rs2[wid] = s2; }
  __syncthreads();
  float S = 0.f, S2 = 0.f;
  #pragma unroll
  for (int i = 0; i < 8; ++i) { S += rs[i]; S2 += rs2[i]; }

  float mean = S / (float)C;
  float var  = (S2 - (float)C * mean * mean) / (float)(C - 1);
  float inv  = rsqrtf(var + 1e-5f);
  for (int i = tid; i < C; i += 256)
    out[(long)row * C + i] = (__bf16)((xr[i] - mean) * inv * w[i] + b[i]);
}

// ---------------------------------------------------------------------------
// Tiled bf16 WMMA GEMM with async-LDS double buffering.
//   out = epilogue(A[M,K] @ W[K,N] + bias [+ res]);  W passed transposed [N][K].
// BM=64, BN=128, BK=64; 256 threads = 8 waves (2x4), wave tile 32x32.
// Staging: 6 x global_load_async_to_lds_b128 per thread per tile pair.
// ---------------------------------------------------------------------------
template <bool GELU_EP, bool RES, bool OUT_BF16>
__global__ __launch_bounds__(256) void gemm_bf16(
    const __bf16* __restrict__ A, const __bf16* __restrict__ WT,
    const float* __restrict__ bias, const float* __restrict__ res,
    float* __restrict__ outF, __bf16* __restrict__ outB,
    int M, int N, int K) {
  constexpr int BM = 64, BN = 128, BK = 64, LDP = BK + 8;
  __shared__ __bf16 As[2][BM][LDP];   //  2 *  64*72*2 = 18 KB
  __shared__ __bf16 Ws[2][BN][LDP];   //  2 * 128*72*2 = 36 KB

  int tid  = threadIdx.x;
  int lane = tid & 31, wid = tid >> 5;
  int waveM = wid >> 2, waveN = wid & 3;      // 2 x 4 wave grid
  int m0 = blockIdx.y * BM;
  int n0 = blockIdx.x * BN;

  // per-stage async staging: A tile 512 chunks, W tile 1024 chunks (16B each)
  auto stage = [&](int buf, int kt) {
    #pragma unroll
    for (int it = 0; it < 2; ++it) {            // A: 64x64
      int idx = tid + it * 256;
      int r = idx >> 3, c = (idx & 7) * 8;
      async_cp16(A + (long)(m0 + r) * K + kt + c, &As[buf][r][c]);
    }
    #pragma unroll
    for (int it = 0; it < 4; ++it) {            // W^T: 128x64
      int idx = tid + it * 256;
      int n = idx >> 3, c = (idx & 7) * 8;
      async_cp16(WT + (long)(n0 + n) * K + kt + c, &Ws[buf][n][c]);
    }
  };

  f32x8 acc[2][2];
  #pragma unroll
  for (int i = 0; i < 2; ++i)
    #pragma unroll
    for (int j = 0; j < 2; ++j) acc[i][j] = (f32x8){};

  stage(0, 0);
  int buf = 0;
  for (int kt = 0; kt < K; kt += BK) {
    if (kt + BK < K) {
      stage(buf ^ 1, kt + BK);   // prefetch next tile while current completes
      wait_async<6>();           // 6 outstanding = the batch just issued
    } else {
      wait_async<0>();
    }
    __syncthreads();

    #pragma unroll
    for (int kk = 0; kk < BK; kk += 32) {
      bf16x16 af[2], bfr[2];
      #pragma unroll
      for (int mt = 0; mt < 2; ++mt)
        af[mt] = load_frag_a_lds(&As[buf][0][0], waveM * 32 + mt * 16, LDP, kk, lane);
      #pragma unroll
      for (int nt = 0; nt < 2; ++nt)
        bfr[nt] = load_frag_b_ldsT(&Ws[buf][0][0], waveN * 32 + nt * 16, LDP, kk, lane);
      #pragma unroll
      for (int mt = 0; mt < 2; ++mt)
        #pragma unroll
        for (int nt = 0; nt < 2; ++nt)
          acc[mt][nt] = wmma_bf16(af[mt], bfr[nt], acc[mt][nt]);
    }
    __syncthreads();
    buf ^= 1;
  }

  // --- epilogue: D layout (VGPR r, lanes0-15 -> M=r, lanes16-31 -> M=r+8) ---
  int nl = lane & 15, hf = lane >> 4;
  #pragma unroll
  for (int mt = 0; mt < 2; ++mt) {
    #pragma unroll
    for (int nt = 0; nt < 2; ++nt) {
      int col = n0 + waveN * 32 + nt * 16 + nl;
      float bc = bias[col];
      #pragma unroll
      for (int r = 0; r < 8; ++r) {
        int row = m0 + waveM * 32 + mt * 16 + r + hf * 8;
        float v = acc[mt][nt][r] + bc;
        if (GELU_EP) v = gelu_tanh(v);
        if (RES)     v += res[(long)row * N + col];
        if (OUT_BF16) outB[(long)row * N + col] = (__bf16)v;
        else          outF[(long)row * N + col] = v;
      }
    }
  }
}

// ---------------------------------------------------------------------------
// Pre-transpose V into vT[b][h][d][t] so P@V B-fragments are contiguous loads
// ---------------------------------------------------------------------------
__global__ void transpose_v(const __bf16* __restrict__ qkv, __bf16* __restrict__ vT) {
  using namespace cfg;
  long total = (long)B * H * D * T;
  long idx = (long)blockIdx.x * blockDim.x + threadIdx.x;
  if (idx >= total) return;
  int t = (int)(idx % T);  long r = idx / T;
  int d = (int)(r % D);    r /= D;
  int h = (int)(r % H);
  int b = (int)(r / H);
  vT[idx] = qkv[((long)(b * T + t)) * C3 + 2 * C + h * D + d];
}

// ---------------------------------------------------------------------------
// Flash-attention: 1 wave per 16-row Q tile, 32 keys/iter, online softmax.
// grid = (T/64, H, B), block = 128 (4 waves).
// ---------------------------------------------------------------------------
__global__ __launch_bounds__(128) void attention_fwd(
    const __bf16* __restrict__ qkv, const __bf16* __restrict__ vT,
    __bf16* __restrict__ out) {
  using namespace cfg;
  int lane = threadIdx.x & 31, wid = threadIdx.x >> 5;
  int h = blockIdx.y, b = blockIdx.z;
  int q0 = blockIdx.x * 64 + wid * 16;

  __shared__ __bf16 Pl[4][16][32];   // per-wave P re-layout buffer (D -> A)

  const __bf16* qbase = qkv + ((long)(b * T + q0)) * C3 + h * D;          // Q rows
  const __bf16* kbase = qkv + ((long)(b * T)) * C3 + C + h * D;           // K rows
  const __bf16* vtb   = vT + ((long)((b * H + h) * D)) * T;               // [d][t]

  bf16x16 qa0 = load_frag_a_rows(qbase, C3, 0, lane);
  bf16x16 qa1 = load_frag_a_rows(qbase, C3, 32, lane);

  f32x8 o[4];
  #pragma unroll
  for (int g = 0; g < 4; ++g) o[g] = (f32x8){};
  float mrun[8], lrun[8];
  #pragma unroll
  for (int r = 0; r < 8; ++r) { mrun[r] = -1e30f; lrun[r] = 0.f; }

  int nl = lane & 15, hf = lane >> 4;
  int kmax = q0 + 15;                       // causal upper bound for this tile

  for (int kt = 0; kt <= kmax; kt += 32) {
    // ---- S = Q @ K^T for 32 keys (two 16-col D tiles) ----
    f32x8 s0 = (f32x8){}, s1 = (f32x8){};
    const __bf16* kb0 = kbase + (long)kt * C3;
    const __bf16* kb1 = kbase + (long)(kt + 16) * C3;
    s0 = wmma_bf16(qa0, load_frag_b_rows(kb0, C3, 0, lane), s0);
    s0 = wmma_bf16(qa1, load_frag_b_rows(kb0, C3, 32, lane), s0);
    s1 = wmma_bf16(qa0, load_frag_b_rows(kb1, C3, 0, lane), s1);
    s1 = wmma_bf16(qa1, load_frag_b_rows(kb1, C3, 32, lane), s1);

    // ---- scale, causal mask, online softmax (rows live in 16-lane groups) ----
    #pragma unroll
    for (int r = 0; r < 8; ++r) {
      int Mrow = r + hf * 8;
      int qrow = q0 + Mrow;
      float a0 = s0[r] * 0.125f; if (kt + nl > qrow)      a0 = -1e30f;
      float a1 = s1[r] * 0.125f; if (kt + 16 + nl > qrow) a1 = -1e30f;
      float mx = fmaxf(a0, a1);
      #pragma unroll
      for (int i = 1; i < 16; i <<= 1) mx = fmaxf(mx, __shfl_xor(mx, i, 32));
      float mnew  = fmaxf(mrun[r], mx);
      float alpha = __expf(mrun[r] - mnew);
      float p0 = __expf(a0 - mnew);
      float p1 = __expf(a1 - mnew);
      float ps = p0 + p1;
      #pragma unroll
      for (int i = 1; i < 16; i <<= 1) ps += __shfl_xor(ps, i, 32);
      lrun[r] = lrun[r] * alpha + ps;
      mrun[r] = mnew;
      #pragma unroll
      for (int g = 0; g < 4; ++g) o[g][r] *= alpha;
      Pl[wid][Mrow][nl]      = (__bf16)p0;    // D layout -> LDS row-major
      Pl[wid][Mrow][16 + nl] = (__bf16)p1;
    }

    // ---- O += P @ V  (P re-read in A layout; V from transposed buffer) ----
    bf16x16 pa = load_frag_a_lds(&Pl[wid][0][0], 0, 32, 0, lane);
    #pragma unroll
    for (int g = 0; g < 4; ++g) {
      bf16x16 vb = load_frag_b_cols(vtb, T, g * 16, kt, lane);
      o[g] = wmma_bf16(pa, vb, o[g]);
    }
  }

  // ---- normalize & store (bf16 attention output, heads concatenated) ----
  #pragma unroll
  for (int r = 0; r < 8; ++r) {
    int row = q0 + r + hf * 8;
    float inv = 1.0f / lrun[r];
    #pragma unroll
    for (int g = 0; g < 4; ++g)
      out[((long)(b * T + row)) * C + h * D + g * 16 + nl] = (__bf16)(o[g][r] * inv);
  }
}

// ---------------------------------------------------------------------------
// Host-side orchestration
// ---------------------------------------------------------------------------
extern "C" void kernel_launch(void* const* d_in, const int* in_sizes, int n_in,
                              void* d_out, int out_size, void* d_ws, size_t ws_size,
                              hipStream_t stream) {
  using namespace cfg;
  (void)in_sizes; (void)n_in; (void)out_size; (void)ws_size;

  const float* x        = (const float*)d_in[0];
  const float* ln1_w    = (const float*)d_in[1];
  const float* ln1_b    = (const float*)d_in[2];
  const float* attn_w   = (const float*)d_in[3];
  const float* attn_b   = (const float*)d_in[4];
  const float* proj_w   = (const float*)d_in[5];
  const float* proj_b   = (const float*)d_in[6];
  const float* ln2_w    = (const float*)d_in[7];
  const float* ln2_b    = (const float*)d_in[8];
  const float* fc_w     = (const float*)d_in[9];
  const float* fc_b     = (const float*)d_in[10];
  const float* fcp_w    = (const float*)d_in[11];
  const float* fcp_b    = (const float*)d_in[12];
  float* out            = (float*)d_out;

  // ---- workspace layout ----
  char* ws = (char*)d_ws;
  size_t off = 0;
  auto take = [&](size_t bytes) -> char* {
    char* p = ws + off;
    off = (off + bytes + 255) & ~(size_t)255;
    return p;
  };
  __bf16* wAttnT = (__bf16*)take((size_t)C * C3 * 2);   // [C3][C]
  __bf16* wProjT = (__bf16*)take((size_t)C * C  * 2);   // [C][C]
  __bf16* wFcT   = (__bf16*)take((size_t)C * C4 * 2);   // [C4][C]
  __bf16* wFcpT  = (__bf16*)take((size_t)C4 * C * 2);   // [C][C4]
  __bf16* h1     = (__bf16*)take((size_t)ROWS * C  * 2);
  __bf16* qkv    = (__bf16*)take((size_t)ROWS * C3 * 2);
  __bf16* vT     = (__bf16*)take((size_t)B * H * D * T * 2);
  __bf16* attnO  = (__bf16*)take((size_t)ROWS * C * 2);
  float*  x1     = (float*) take((size_t)ROWS * C * 4);
  __bf16* h2     = (__bf16*)take((size_t)ROWS * C  * 2);
  __bf16* gact   = (__bf16*)take((size_t)ROWS * C4 * 2);

  auto blocks = [](long n) { return (unsigned)((n + 255) / 256); };

  // ---- 1) transpose + convert weights to bf16 (reused by every M-block) ----
  transpose_w_bf16<<<blocks((long)C  * C3), 256, 0, stream>>>(attn_w, wAttnT, C,  C3);
  transpose_w_bf16<<<blocks((long)C  * C ), 256, 0, stream>>>(proj_w, wProjT, C,  C);
  transpose_w_bf16<<<blocks((long)C  * C4), 256, 0, stream>>>(fc_w,   wFcT,   C,  C4);
  transpose_w_bf16<<<blocks((long)C4 * C ), 256, 0, stream>>>(fcp_w,  wFcpT,  C4, C);

  // ---- 2) ln1 ----
  layernorm_bf16<<<ROWS, 256, 0, stream>>>(x, ln1_w, ln1_b, h1);

  // ---- 3) qkv = h1 @ attn_w + attn_b  (bf16 out) ----
  gemm_bf16<false, false, true><<<dim3(C3 / 128, ROWS / 64), 256, 0, stream>>>(
      h1, wAttnT, attn_b, nullptr, nullptr, qkv, ROWS, C3, C);

  // ---- 4) transpose V ----
  transpose_v<<<blocks((long)B * H * D * T), 256, 0, stream>>>(qkv, vT);

  // ---- 5) causal flash attention ----
  attention_fwd<<<dim3(T / 64, H, B), 128, 0, stream>>>(qkv, vT, attnO);

  // ---- 6) x1 = x + attnO @ proj_w + proj_b  (f32 out) ----
  gemm_bf16<false, true, false><<<dim3(C / 128, ROWS / 64), 256, 0, stream>>>(
      attnO, wProjT, proj_b, x, x1, nullptr, ROWS, C, C);

  // ---- 7) ln2 ----
  layernorm_bf16<<<ROWS, 256, 0, stream>>>(x1, ln2_w, ln2_b, h2);

  // ---- 8) g = gelu(h2 @ fc_w + fc_b)  (bf16 out) ----
  gemm_bf16<true, false, true><<<dim3(C4 / 128, ROWS / 64), 256, 0, stream>>>(
      h2, wFcT, fc_b, nullptr, nullptr, gact, ROWS, C4, C);

  // ---- 9) out = x1 + g @ fc_proj_w + fc_proj_b  (f32 out) ----
  gemm_bf16<false, true, false><<<dim3(C / 128, ROWS / 64), 256, 0, stream>>>(
      gact, wFcpT, fcp_b, x1, out, nullptr, ROWS, C, C4);
}
